// MethodName_85186381349630
// MI455X (gfx1250) — compile-verified, matrix operand order checked by reading, not run on hardware
//
#include <hip/hip_runtime.h>
#include <hip/hip_fp16.h>

#define NN 2048
#define CC 128
#define BS 8
#define NUM_LAYER 6

typedef __attribute__((ext_vector_type(16))) _Float16 v16h;
typedef __attribute__((ext_vector_type(8)))  _Float16 h8;
typedef __attribute__((ext_vector_type(8)))  float    v8f;

// ---------------- K1: Q/K projections (tiny GEMM, scalar) ----------------
// out[(b*N+n)*C + c] = bias[c] + sum_k W[c,k] * feat[b,k,n]   (f16 output)
__global__ void proj_kernel(const float* __restrict__ W, const float* __restrict__ bias,
                            const float* __restrict__ feat, _Float16* __restrict__ out)
{
    int bn = blockIdx.x;                 // b*N + n
    int b  = bn >> 11;
    int n  = bn & (NN - 1);
    int c  = threadIdx.x;                // 0..127
    const float* frow = feat + (size_t)b * CC * NN + n;   // stride NN over k (uniform per block)
    const float* wrow = W + (size_t)c * CC;
    float acc = bias[c];
#pragma unroll 4
    for (int k = 0; k < CC; ++k)
        acc += wrow[k] * frow[(size_t)k * NN];
    out[(size_t)bn * CC + c] = (_Float16)acc;
}

// ---------------- K2: attention GEMM (WMMA) + mask + sigmoid ----------------
// score[b,o,i] = sigmoid(att/sqrt(128)) if H0>0 else 0 ; staged into the W output region
__global__ void attn_kernel(const _Float16* __restrict__ Qt, const _Float16* __restrict__ Kt,
                            const float* __restrict__ H0, float* __restrict__ score)
{
    int b    = blockIdx.z;
    int wave = threadIdx.x >> 5;
    int lane = threadIdx.x & 31;
    int o0 = blockIdx.x * 64 + (wave >> 1) * 16;
    int i0 = blockIdx.y * 32 + (wave & 1) * 16;

    int r    = lane & 15;
    int koff = (lane >> 4) * 8;          // ISA 16-bit A layout: lanes16-31 start at K=8

    const _Float16* aBase = Qt + ((size_t)b * NN + o0 + r) * CC + koff;
    const _Float16* bBase = Kt + ((size_t)b * NN + i0 + r) * CC + koff;

    v8f acc = {};
#pragma unroll
    for (int kc = 0; kc < CC; kc += 32) {
        h8 alo = *(const h8*)(aBase + kc);
        h8 ahi = *(const h8*)(aBase + kc + 16);
        h8 blo = *(const h8*)(bBase + kc);
        h8 bhi = *(const h8*)(bBase + kc + 16);
        v16h a, bb;
#pragma unroll
        for (int e = 0; e < 8; ++e) {
            a[e] = alo[e];  a[e + 8] = ahi[e];
            bb[e] = blo[e]; bb[e + 8] = bhi[e];
        }
        acc = __builtin_amdgcn_wmma_f32_16x16x32_f16(false, a, false, bb,
                                                     (short)0, acc, false, false);
    }

    const float scale = 0.08838834764831845f;  // 1/sqrt(128)
    int i     = i0 + (lane & 15);
    int obase = o0 + ((lane >> 4) << 3);       // C/D layout: lanes16-31 hold M=v+8
#pragma unroll
    for (int v = 0; v < 8; ++v) {
        int o = obase + v;
        size_t idx = ((size_t)b * NN + o) * NN + i;
        float m = H0[idx];
        // branch-free sigmoid: native rcp (v_rcp_f32), then select on mask
        float x = acc[v] * scale;
        float e = __expf(-x);
        float s = __builtin_amdgcn_rcpf(1.0f + e);
        s = (m > 0.0f) ? s : 0.0f;
        score[idx] = s;
    }
}

// ---------------- K3: exact per-row k-th largest via MSB-first radix select ----------------
__global__ void topk_kernel(const float* __restrict__ score, float* __restrict__ amin,
                            const int* __restrict__ itp)
{
    __shared__ unsigned hist[256];
    __shared__ unsigned sel, rem_s;
    int it = itp[0];
    unsigned k = (unsigned)(int)rintf((float)NN * 0.1f * (float)(NUM_LAYER - 1 - it));
    int row = blockIdx.x;                       // b*N + o
    const float* p = score + (size_t)row * NN;
    int t = threadIdx.x;
    unsigned prefix = 0, remaining = k;

    for (int pass = 0; pass < 4; ++pass) {
        int shift = 24 - 8 * pass;
        hist[t] = 0;
        __syncthreads();
        unsigned hmask = (pass == 0) ? 0u : (0xFFFFFFFFu << (shift + 8));
        for (int j = t; j < NN; j += 256) {
            unsigned u = __float_as_uint(p[j]);   // scores >= 0 -> bits order-preserving
            if (((u ^ prefix) & hmask) == 0)
                atomicAdd(&hist[(u >> shift) & 255], 1u);
        }
        __syncthreads();
        if (t == 0) {
            unsigned cum = 0, d = 0, rr = remaining;
            for (int dd = 255; dd >= 0; --dd) {
                unsigned c = hist[dd];
                if (cum + c >= remaining) { d = (unsigned)dd; rr = remaining - cum; break; }
                cum += c;
            }
            sel = d; rem_s = rr;
        }
        __syncthreads();
        prefix |= (sel << shift);
        remaining = rem_s;
        __syncthreads();
    }
    if (t == 0) amin[row] = __uint_as_float(prefix);
}

// ---------------- K4: threshold in place, build H, row sums (deg_V) ----------------
__global__ void thresh_kernel(const float* __restrict__ H0, const float* __restrict__ amin,
                              float* __restrict__ Hout, float* __restrict__ Wout,
                              float* __restrict__ degV)
{
    __shared__ float red[256];
    int row = blockIdx.x;                       // b*N + o
    float a = amin[row];
    size_t base = (size_t)row * NN;
    float local = 0.0f;
    for (int j = threadIdx.x; j < NN; j += 256) {
        float s = Wout[base + j];
        float w = (s >= a) ? s : 0.0f;
        Wout[base + j] = w;
        float h = (w > 0.0f) ? H0[base + j] : 0.0f;
        Hout[base + j] = h;
        local += h;
    }
    red[threadIdx.x] = local;
    __syncthreads();
    for (int s = 128; s > 0; s >>= 1) {
        if (threadIdx.x < s) red[threadIdx.x] += red[threadIdx.x + s];
        __syncthreads();
    }
    if (threadIdx.x == 0) degV[row] = red[0];
}

// ---------------- K5: column sums (deg_E, W column-sum), coalesced over i ----------------
__global__ void colsum_kernel(const float* __restrict__ Hout, const float* __restrict__ Wout,
                              float* __restrict__ degE, float* __restrict__ wcol)
{
    int gid = blockIdx.x * 256 + threadIdx.x;   // b*N + i
    int b = gid >> 11;
    int i = gid & (NN - 1);
    size_t base = (size_t)b * NN * NN + i;
    float de = 0.0f, wc = 0.0f;
    for (int o = 0; o < NN; ++o) {
        de += Hout[base + (size_t)o * NN];
        wc += Wout[base + (size_t)o * NN];
    }
    degE[gid] = de;
    wcol[gid] = wc;
}

// ---------------- K6: fill De_n_1, Dv_n_1 (zero + diagonal), float4 stores ----------------
__global__ void diag_kernel(const float* __restrict__ degE, const float* __restrict__ degV,
                            float* __restrict__ De, float* __restrict__ Dv)
{
    size_t tid = (size_t)blockIdx.x * 256 + threadIdx.x;  // BS*N*(N/4) threads
    int    i4  = (int)(tid & (NN / 4 - 1));
    size_t row = tid >> 9;                                // b*N + o
    int    o   = (int)(row & (NN - 1));
    int    i0  = i4 * 4;

    float4 ze = {0.f, 0.f, 0.f, 0.f};
    float4 zv = {0.f, 0.f, 0.f, 0.f};
    if (o >= i0 && o < i0 + 4) {
        float de = degE[row];
        float dv = degV[row];
        ((float*)&ze)[o - i0] = (de != 0.0f) ? 1.0f / de : 0.0f;
        ((float*)&zv)[o - i0] = (dv != 0.0f) ? 1.0f / dv : 0.0f;
    }
    *(float4*)(De + row * NN + i0) = ze;
    *(float4*)(Dv + row * NN + i0) = zv;
}

// ---------------- K7: W_edge = colsum / max(||colsum||2, 1e-12), per batch ----------------
__global__ void wedge_kernel(const float* __restrict__ wcol, float* __restrict__ Wedge)
{
    __shared__ float red[256];
    int b = blockIdx.x;
    const float* p = wcol + (size_t)b * NN;
    float local = 0.0f;
    for (int j = threadIdx.x; j < NN; j += 256) { float v = p[j]; local += v * v; }
    red[threadIdx.x] = local;
    __syncthreads();
    for (int s = 128; s > 0; s >>= 1) {
        if (threadIdx.x < s) red[threadIdx.x] += red[threadIdx.x + s];
        __syncthreads();
    }
    float nrm = fmaxf(sqrtf(red[0]), 1e-12f);
    for (int j = threadIdx.x; j < NN; j += 256)
        Wedge[(size_t)b * NN + j] = p[j] / nrm;
}

extern "C" void kernel_launch(void* const* d_in, const int* in_sizes, int n_in,
                              void* d_out, int out_size, void* d_ws, size_t ws_size,
                              hipStream_t stream) {
    const float* H0     = (const float*)d_in[0];
    const float* vertex = (const float*)d_in[1];
    const float* edge   = (const float*)d_in[2];
    const float* Wq     = (const float*)d_in[3];
    const float* bq     = (const float*)d_in[4];
    const float* Wk     = (const float*)d_in[5];
    const float* bk     = (const float*)d_in[6];
    const int*   itp    = (const int*)d_in[7];

    float* out = (float*)d_out;
    const size_t S = (size_t)BS * NN * NN;
    float* Hout  = out;
    float* Wout  = out + S;        // staged: score -> thresholded W (in place)
    float* De    = out + 2 * S;
    float* Dv    = out + 3 * S;
    float* Wedge = out + 4 * S;

    _Float16* Qt  = (_Float16*)d_ws;                 // BS*N*C halves
    _Float16* Kt  = Qt + (size_t)BS * NN * CC;
    float* amin   = (float*)(Kt + (size_t)BS * NN * CC);
    float* degV   = amin + (size_t)BS * NN;
    float* degE   = degV + (size_t)BS * NN;
    float* wcol   = degE + (size_t)BS * NN;

    // K1: projections
    proj_kernel<<<BS * NN, CC, 0, stream>>>(Wq, bq, vertex, Qt);
    proj_kernel<<<BS * NN, CC, 0, stream>>>(Wk, bk, edge, Kt);

    // K2: WMMA attention + mask + sigmoid -> Wout (score staging)
    dim3 agrid(NN / 64, NN / 32, BS);
    attn_kernel<<<agrid, 256, 0, stream>>>(Qt, Kt, H0, Wout);

    // K3: exact k-th largest per row
    topk_kernel<<<BS * NN, 256, 0, stream>>>(Wout, amin, itp);

    // K4: threshold in place, H, deg_V
    thresh_kernel<<<BS * NN, 256, 0, stream>>>(H0, amin, Hout, Wout, degV);

    // K5: deg_E and W column sums
    colsum_kernel<<<(BS * NN) / 256, 256, 0, stream>>>(Hout, Wout, degE, wcol);

    // K6: De_n_1 / Dv_n_1 diagonal matrices
    diag_kernel<<<(unsigned)((S / 4) / 256), 256, 0, stream>>>(degE, degV, De, Dv);

    // K7: W_edge
    wedge_kernel<<<BS, 256, 0, stream>>>(wcol, Wedge);
}
